// CrossViewTransformer_69861938037560
// MI455X (gfx1250) — matrix operand order repaired and checked by reading, not compiled, not run on hardware
//
#include <hip/hip_runtime.h>
#include <hip/hip_bf16.h>
#include <stdint.h>

typedef __attribute__((ext_vector_type(16))) __bf16 v16bf;
typedef __attribute__((ext_vector_type(8)))  float  v8f;
typedef __attribute__((ext_vector_type(2)))  float  v2f;

static __device__ __forceinline__ unsigned short f2bf(float f) {
  union { float f; unsigned u; } x; x.f = f;
  unsigned r = x.u + 0x7FFFu + ((x.u >> 16) & 1u);   // round-to-nearest-even
  return (unsigned short)(r >> 16);
}
static __device__ __forceinline__ __bf16 us2bf(unsigned short u) {
  union { unsigned short s; __bf16 b; } x; x.s = u; return x.b;
}

// ---------------- elementwise prep kernels ----------------

// f32 [BC,Hs,Ws] -> bf16 padded [BC,Hs+2,Ws+2]; reflect=1: mirror pad, else zero pad
__global__ __launch_bounds__(256) void pad_bf16_k(const float* __restrict__ in,
                                                  unsigned short* __restrict__ out,
                                                  int BC, int Hs, int Ws, int reflect) {
  const int Hp = Hs + 2, Wp = Ws + 2;
  long t = (long)blockIdx.x * blockDim.x + threadIdx.x;
  long total = (long)BC * Hp * Wp;
  if (t >= total) return;
  int xo = t % Wp; long r = t / Wp; int yo = r % Hp; long bc = r / Hp;
  int xi = xo - 1, yi = yo - 1;
  float val = 0.f;
  if (reflect) {
    xi = (xi < 0) ? 1 : (xi >= Ws ? Ws - 2 : xi);
    yi = (yi < 0) ? 1 : (yi >= Hs ? Hs - 2 : yi);
    val = in[(bc * Hs + yi) * Ws + xi];
  } else if (xi >= 0 && xi < Ws && yi >= 0 && yi < Hs) {
    val = in[(bc * Hs + yi) * Ws + xi];
  }
  out[t] = f2bf(val);
}

// weights [O,Ci,3,3] f32 -> fragment-ready bf16.
// K-order: k = (kh*3+kw)*Ci + ci.  Layout: slot = nt*KB + kb (KB = Ci*9/32),
// element t = (slot*32 + lane)*16 + e holds B-fragment value for that lane:
// ln=lane&15, g=lane>>4 -> k = kb*32 + 8g + (e&7) + 16*(e>>3), n = nt*16+ln
__global__ __launch_bounds__(256) void prep_wswz_k(const float* __restrict__ w,
                                                   unsigned short* __restrict__ ws,
                                                   int Ci, int Co) {
  long t = (long)blockIdx.x * blockDim.x + threadIdx.x;
  long total = (long)Ci * 9 * Co;
  if (t >= total) return;
  int e = t & 15; int lane = (t >> 4) & 31; long slot = t >> 9;
  int KB = (Ci * 9) >> 5;
  int nt = slot / KB, kb = slot % KB;
  int ln = lane & 15, g = lane >> 4;
  int v = e >> 1, p = e & 1;
  int k = kb * 32 + 8 * g + (e & 7) + ((e >> 3) << 4);
  int r9 = k / Ci, ci = k - r9 * Ci;
  int kh = r9 / 3, kw = r9 - kh * 3;
  int o = nt * 16 + ln;
  ws[t] = f2bf(w[(((long)o * Ci + ci) * 3 + kh) * 3 + kw]);
}

__global__ __launch_bounds__(256) void maxpool2_k(const float* __restrict__ in,
                                                  float* __restrict__ outf,
                                                  int BC, int Hs, int Ws) {
  int Ho = Hs >> 1, Wo = Ws >> 1;
  long t = (long)blockIdx.x * blockDim.x + threadIdx.x;
  long total = (long)BC * Ho * Wo;
  if (t >= total) return;
  int wo = t % Wo; long tmp = t / Wo; int ho = tmp % Ho; long bc = tmp / Ho;
  const float* ib = in + ((long)bc * Hs + 2 * ho) * Ws + 2 * wo;
  outf[t] = fmaxf(fmaxf(ib[0], ib[1]), fmaxf(ib[Ws], ib[Ws + 1]));
}

// T[b,c,m] = v[b,c, amax[b,m]]
__global__ __launch_bounds__(256) void gather_T_k(const float* __restrict__ v,
                                                  const int* __restrict__ amax,
                                                  float* __restrict__ T,
                                                  int Cc, int Nsp, long total) {
  long t = (long)blockIdx.x * blockDim.x + threadIdx.x;
  if (t >= total) return;
  int m = t % Nsp; long bc = t / Nsp; long b = bc / Cc;
  T[t] = v[bc * Nsp + amax[b * Nsp + m]];
}

// cat_pad bf16 [B, 2C, 34, 34] zero-padded: first C channels front_x, next C = T
__global__ __launch_bounds__(256) void pack_concat_pad_k(const float* __restrict__ fx,
                                                         const float* __restrict__ T,
                                                         unsigned short* __restrict__ cat,
                                                         long total) {
  long t = (long)blockIdx.x * blockDim.x + threadIdx.x;
  if (t >= total) return;
  int xo = t % 34; long r = t / 34; int yo = r % 34; r /= 34;
  int c = r % 256; long b = r / 256;
  int xi = xo - 1, yi = yo - 1;
  float val = 0.f;
  if (xi >= 0 && xi < 32 && yi >= 0 && yi < 32) {
    long sp = yi * 32 + xi;
    val = (c < 128) ? fx[((long)b * 128 + c) * 1024 + sp]
                    : T[((long)b * 128 + (c - 128)) * 1024 + sp];
  }
  cat[t] = f2bf(val);
}

// ---------------- WMMA kernels ----------------

// implicit-GEMM 3x3 conv over pre-padded bf16 input with fragment-ready weights.
// K-order k = (kh*3+kw)*CIN + ci -> tap index is scalar per K block, channel
// offsets are compile-time constants. One wave computes 16 spatial x 64 Cout.
// grid.x = B*(HS*WS/16), grid.y = COUT/64.
template <int CIN, int COUT, int HS, int WS>
__global__ __launch_bounds__(32) void conv3x3_wmma_k(const unsigned short* __restrict__ in_pad,
                                                     const unsigned short* __restrict__ wswz,
                                                     const float* __restrict__ bias,
                                                     float* __restrict__ out) {
  constexpr int HW = HS * WS, WP = WS + 2, HWP = (HS + 2) * (WS + 2);
  constexpr int KB = (CIN * 9) / 32;
  constexpr int TPI = HW / 16;
  const int lane = threadIdx.x, g = lane >> 4, ln = lane & 15;
  const int b = blockIdx.x / TPI;
  const int mbase = (blockIdx.x % TPI) << 4;
  const int nt0 = blockIdx.y << 2;
  const int m = mbase + ln;
  const int y0 = m / WS, x0 = m % WS;           // WS is a power of two -> shifts
  const unsigned short* inb = in_pad + (long)b * CIN * HWP;
  const v16bf* wv = (const v16bf*)wswz;
  v8f acc[4] = {};
  int kb = 0;
#pragma unroll
  for (int r9 = 0; r9 < 9; ++r9) {
    const int kh = r9 / 3, kw = r9 - (r9 / 3) * 3;
    const unsigned short* ap = inb + (y0 + kh) * WP + (x0 + kw) + (long)(8 * g) * HWP;
    for (int cc = 0; cc < CIN; cc += 32, ++kb, ap += (long)32 * HWP) {
      v16bf a;
#pragma unroll
      for (int v = 0; v < 8; ++v) {
        const int c0 = (v < 4) ? (2 * v) : (8 + 2 * v);   // A layout K offsets
        a[2 * v]     = us2bf(ap[(long)c0 * HWP]);
        a[2 * v + 1] = us2bf(ap[(long)(c0 + 1) * HWP]);
      }
#pragma unroll
      for (int q = 0; q < 4; ++q) {
        v16bf bfr = wv[((long)(nt0 + q) * KB + kb) * 32 + lane];  // 32B aligned
        acc[q] = __builtin_amdgcn_wmma_f32_16x16x32_bf16(false, a, false, bfr,
                                                         (short)0, acc[q], false, false);
      }
    }
  }
#pragma unroll
  for (int q = 0; q < 4; ++q) {                  // D: M = r + 8g (consecutive m)
    int n = (nt0 + q) * 16 + ln;
    float bi = bias[n];
    float* ob = out + ((long)b * COUT + n) * HW + mbase + 8 * g;
    float4 s0 = {acc[q][0] + bi, acc[q][1] + bi, acc[q][2] + bi, acc[q][3] + bi};
    float4 s1 = {acc[q][4] + bi, acc[q][5] + bi, acc[q][6] + bi, acc[q][7] + bi};
    *(float4*)ob = s0;
    *(float4*)(ob + 4) = s1;
  }
}

// 1x1 conv as GEMM (exact f32 WMMA): out[b,o,m] = sum_c W[o,c] x[b,c,m] + bias[o]
// grid.x = B*(Nsp/16), grid.y = CO/16
template <int CI, int CO>
__global__ __launch_bounds__(32) void conv1x1_wmma_k(const float* __restrict__ x,
                                                     const float* __restrict__ Wm,
                                                     const float* __restrict__ bias,
                                                     float* __restrict__ out, int Nsp) {
  const int lane = threadIdx.x, g = lane >> 4, ln = lane & 15;
  const int tpi = Nsp >> 4;
  const int b = blockIdx.x / tpi, mbase = (blockIdx.x % tpi) << 4;
  const int ntile = blockIdx.y;
  const int m = mbase + ln;
  const float* xb = x + (long)b * CI * Nsp;
  const float* wb = Wm + (long)(ntile * 16 + ln) * CI;
  v8f acc = {};
  for (int kk = 0; kk < CI; kk += 4) {
    v2f a, bv;
#pragma unroll
    for (int r = 0; r < 2; ++r) {
      int c = kk + r + 2 * g;                     // f32 A/B layout: K = r + 2g
      a[r]  = xb[(long)c * Nsp + m];
      bv[r] = wb[c];
    }
    acc = __builtin_amdgcn_wmma_f32_16x16x4_f32(false, a, false, bv,
                                                (short)0, acc, false, false);
  }
  const int o = ntile * 16 + ln;
  const float bi = bias[o];
  float* ob = out + ((long)b * CO + o) * Nsp + mbase + 8 * g;
  float4 s0 = {acc[0] + bi, acc[1] + bi, acc[2] + bi, acc[3] + bi};
  float4 s1 = {acc[4] + bi, acc[5] + bi, acc[6] + bi, acc[7] + bi};
  *(float4*)ob = s0;
  *(float4*)(ob + 4) = s1;
}

// fused energy + column max/argmax: energy[i,j] = sum_c k[b,c,i] q[b,c,j] (K=16)
// per wave: one 16-wide j tile; loop i tiles; never materialize energy.
// Branchless running (max, argmax) update: first-occurrence semantics like
// jnp.argmax (on ties keep the smaller i), no exec-mask manipulation.
__global__ __launch_bounds__(32) void energy_max_k(const float* __restrict__ kbuf,
                                                   const float* __restrict__ qbuf,
                                                   float* __restrict__ maxout,
                                                   float* __restrict__ maxout2,
                                                   int* __restrict__ argout, int Nsp) {
  const int lane = threadIdx.x, g = lane >> 4, ln = lane & 15;
  const int tpi = Nsp >> 4;
  const int b = blockIdx.x / tpi, jbase = (blockIdx.x % tpi) << 4;
  const int j = jbase + ln;
  const float* kb = kbuf + (long)b * 16 * Nsp;
  const float* qb = qbuf + (long)b * 16 * Nsp;
  float best = -__builtin_inff(); int besti = 0;
  for (int it = 0; it < tpi; ++it) {
    v8f acc = {};
#pragma unroll
    for (int kk = 0; kk < 16; kk += 4) {
      v2f a, bv;
#pragma unroll
      for (int r = 0; r < 2; ++r) {
        int c = kk + r + 2 * g;
        a[r]  = kb[(long)c * Nsp + it * 16 + ln];  // A: M = i(lane%16), K = c
        bv[r] = qb[(long)c * Nsp + j];             // B: N = j(lane%16), K = c
      }
      acc = __builtin_amdgcn_wmma_f32_16x16x4_f32(false, a, false, bv,
                                                  (short)0, acc, false, false);
    }
#pragma unroll
    for (int r = 0; r < 8; ++r) {                  // D: i = it*16 + r + 8g
      int i = it * 16 + r + 8 * g;
      float v = acc[r];
      bool better = (v > best) | ((v == best) & (i < besti));   // no short-circuit
      best  = better ? v : best;
      besti = better ? i : besti;
    }
    float ob = __shfl_xor(best, 16, 32);           // combine the two lane halves
    int   oi = __shfl_xor(besti, 16, 32);
    bool better = (ob > best) | ((ob == best) & (oi < besti));
    best  = better ? ob : best;
    besti = better ? oi : besti;
  }
  if (g == 0) {
    maxout[(long)b * Nsp + j] = best;
    if (maxout2) maxout2[(long)b * Nsp + j] = best;
    if (argout)  argout[(long)b * Nsp + j] = besti;
  }
}

// epilogue: out = front_x + front_res*S + attn_mat(32x32) @ vd(32x32) per (b,c)
__global__ __launch_bounds__(32) void xdepth_epilogue_k(const float* __restrict__ attn,
                                                        const float* __restrict__ vd,
                                                        const float* __restrict__ fx,
                                                        const float* __restrict__ fres,
                                                        const float* __restrict__ S,
                                                        float* __restrict__ out) {
  const int idx = blockIdx.x;
  const int nt = idx & 1, mt = (idx >> 1) & 1;
  const long bc = idx >> 2;
  const long b = bc >> 7;
  const int lane = threadIdx.x, g = lane >> 4, ln = lane & 15;
  const float* ab = attn + b * 1024;
  const float* vb = vd + bc * 1024;
  v8f acc = {};
#pragma unroll
  for (int kk = 0; kk < 32; kk += 4) {
    v2f a, bv;
#pragma unroll
    for (int r = 0; r < 2; ++r) {
      int k = kk + r + 2 * g;
      a[r]  = ab[(mt * 16 + ln) * 32 + k];         // A[w,k]
      bv[r] = vb[k * 32 + nt * 16 + ln];           // B[k,h]
    }
    acc = __builtin_amdgcn_wmma_f32_16x16x4_f32(false, a, false, bv,
                                                (short)0, acc, false, false);
  }
  const float* fxb = fx + bc * 1024;
  const float* frb = fres + bc * 1024;
  const float* Sb  = S + b * 1024;
  float* ob = out + bc * 1024;
#pragma unroll
  for (int r = 0; r < 8; ++r) {
    int w = mt * 16 + r + 8 * g, h = nt * 16 + ln;
    int sp = w * 32 + h;
    ob[sp] = fxb[sp] + frb[sp] * Sb[sp] + acc[r];
  }
}

// ---------------- host launch ----------------

extern "C" void kernel_launch(void* const* d_in, const int* in_sizes, int n_in,
                              void* d_out, int out_size, void* d_ws, size_t ws_size,
                              hipStream_t stream) {
  const int B = 8, C = 128, Nsp = 1024;            // 32x32 spatial
  const float* front_x     = (const float*)d_in[0];
  const float* cross_x     = (const float*)d_in[1];
  const float* front_x_hat = (const float*)d_in[2];
  const float* dfeat       = (const float*)d_in[3];
  const float *Wq=(const float*)d_in[4],  *bq=(const float*)d_in[5];
  const float *Wk=(const float*)d_in[6],  *bk=(const float*)d_in[7];
  const float *Wv=(const float*)d_in[8],  *bv=(const float*)d_in[9];
  const float *Wf=(const float*)d_in[10], *bf=(const float*)d_in[11];
  const float *Wqd=(const float*)d_in[12],*bqd=(const float*)d_in[13];
  const float *Wkd=(const float*)d_in[14],*bkd=(const float*)d_in[15];
  const float *Wvd=(const float*)d_in[16],*bvd=(const float*)d_in[17];
  const float *Wc1=(const float*)d_in[18],*bc1=(const float*)d_in[19];
  const float *Wc2=(const float*)d_in[20],*bc2=(const float*)d_in[21];

  float* outmain = (float*)d_out;                  // [B,128,32,32]
  float* Sout    = outmain + (long)B * C * Nsp;    // [B,1,32,32]
  float* attnout = Sout + (long)B * Nsp;           // [B,1,32,32]

  uint8_t* w8 = (uint8_t*)d_ws;
  size_t off = 0;
  auto take = [&](size_t bytes) -> void* {
    void* p = w8 + off;
    off += (bytes + 255) & ~(size_t)255;
    return p;
  };
  unsigned short* dpad  = (unsigned short*)take((size_t)B*512*130*130*2);
  unsigned short* ws1   = (unsigned short*)take((size_t)512*9*128*2);
  float*          c1out = (float*)take((size_t)B*128*128*128*4);
  float*          p1f   = (float*)take((size_t)B*128*64*64*4);
  unsigned short* p1pad = (unsigned short*)take((size_t)B*128*66*66*2);
  unsigned short* ws2   = (unsigned short*)take((size_t)128*9*128*2);
  float*          c2out = (float*)take((size_t)B*128*64*64*4);
  float*          dbuf  = (float*)take((size_t)B*128*32*32*4);
  float*          qb    = (float*)take((size_t)B*16*Nsp*4);
  float*          kb    = (float*)take((size_t)B*16*Nsp*4);
  float*          qdb   = (float*)take((size_t)B*16*Nsp*4);
  float*          kdb   = (float*)take((size_t)B*16*Nsp*4);
  float*          vbuf  = (float*)take((size_t)B*C*Nsp*4);
  float*          vdb   = (float*)take((size_t)B*C*Nsp*4);
  float*          fstar = (float*)take((size_t)B*Nsp*4);
  int*            amax  = (int*)take((size_t)B*Nsp*4);
  float*          Tb    = (float*)take((size_t)B*C*Nsp*4);
  unsigned short* catp  = (unsigned short*)take((size_t)B*256*34*34*2);
  unsigned short* wsf   = (unsigned short*)take((size_t)256*9*128*2);
  float*          fres  = (float*)take((size_t)B*C*Nsp*4);

  // ---- depth branch: pad+conv3x3(reflect) -> pool -> pad+conv3x3(reflect) -> pool
  {
    long n = (long)B * 512 * 130 * 130;
    pad_bf16_k<<<(unsigned)((n + 255) / 256), 256, 0, stream>>>(dfeat, dpad, B * 512, 128, 128, 1);
  }
  prep_wswz_k<<<(512*9*128 + 255) / 256, 256, 0, stream>>>(Wc1, ws1, 512, 128);
  prep_wswz_k<<<(128*9*128 + 255) / 256, 256, 0, stream>>>(Wc2, ws2, 128, 128);
  prep_wswz_k<<<(256*9*128 + 255) / 256, 256, 0, stream>>>(Wf,  wsf, 256, 128);

  conv3x3_wmma_k<512,128,128,128><<<dim3(B * (128*128/16), 2), 32, 0, stream>>>(dpad, ws1, bc1, c1out);
  maxpool2_k<<<(unsigned)(((long)B*128*64*64 + 255) / 256), 256, 0, stream>>>(c1out, p1f, B*128, 128, 128);
  {
    long n = (long)B * 128 * 66 * 66;
    pad_bf16_k<<<(unsigned)((n + 255) / 256), 256, 0, stream>>>(p1f, p1pad, B * 128, 64, 64, 1);
  }
  conv3x3_wmma_k<128,128,64,64><<<dim3(B * (64*64/16), 2), 32, 0, stream>>>(p1pad, ws2, bc2, c2out);
  maxpool2_k<<<(unsigned)(((long)B*128*32*32 + 255) / 256), 256, 0, stream>>>(c2out, dbuf, B*128, 64, 64);

  // ---- 1x1 convs (exact f32 WMMA) ----
  conv1x1_wmma_k<128,16> <<<dim3(B*(Nsp/16), 1), 32, 0, stream>>>(cross_x,     Wq,  bq,  qb,  Nsp);
  conv1x1_wmma_k<128,16> <<<dim3(B*(Nsp/16), 1), 32, 0, stream>>>(front_x,     Wk,  bk,  kb,  Nsp);
  conv1x1_wmma_k<128,128><<<dim3(B*(Nsp/16), 8), 32, 0, stream>>>(front_x_hat, Wv,  bv,  vbuf,Nsp);
  conv1x1_wmma_k<128,16> <<<dim3(B*(Nsp/16), 1), 32, 0, stream>>>(cross_x,     Wqd, bqd, qdb, Nsp);
  conv1x1_wmma_k<128,16> <<<dim3(B*(Nsp/16), 1), 32, 0, stream>>>(front_x,     Wkd, bkd, kdb, Nsp);
  conv1x1_wmma_k<128,128><<<dim3(B*(Nsp/16), 8), 32, 0, stream>>>(dbuf,        Wvd, bvd, vdb, Nsp);

  // ---- fused energy + max/argmax (WMMA f32, no 1024x1024 materialization) ----
  energy_max_k<<<B * (Nsp/16), 32, 0, stream>>>(kb,  qb,  fstar,   Sout,    amax,    Nsp);
  energy_max_k<<<B * (Nsp/16), 32, 0, stream>>>(kdb, qdb, attnout, nullptr, nullptr, Nsp);

  // ---- gather T, concat(+zero pad), front conv, fused epilogue ----
  {
    long n = (long)B * C * Nsp;
    gather_T_k<<<(unsigned)((n + 255) / 256), 256, 0, stream>>>(vbuf, amax, Tb, C, Nsp, n);
  }
  {
    long n = (long)B * 256 * 34 * 34;
    pack_concat_pad_k<<<(unsigned)((n + 255) / 256), 256, 0, stream>>>(front_x, Tb, catp, n);
  }
  conv3x3_wmma_k<256,128,32,32><<<dim3(B * (Nsp/16), 2), 32, 0, stream>>>(catp, wsf, bf, fres);

  xdepth_epilogue_k<<<B * C * 4, 32, 0, stream>>>(attnout, vdb, front_x, fres, fstar, outmain);
}
